// MRT_phi_18511309046361
// MI455X (gfx1250) — matrix operand order repaired and checked by reading, not compiled, not run on hardware
//
#include <hip/hip_runtime.h>
#include <hip/hip_bf16.h>
#include <math.h>

// ---------------- problem constants ----------------
#define NN 8192
#define BB 8
#define SS 1024
#define EE 65536
#define E2 73728      // EE + NN self loops
#define DIN 4
#define EMB 64
#define THD 4
#define HD 16
#define DFF 256
#define TOUT 16
#define FEATC 21
#define FKP 32        // FEATC padded to 32
#define HID 512
#define GH 6
#define GCOL 3072     // GH*HID
#define INFK 513
#define INFKP 544     // padded to mult of 32
#define NUM_LOOP 3

typedef __bf16 bf16_t;
typedef bf16_t v16bf __attribute__((ext_vector_type(16)));
typedef float  v8f   __attribute__((ext_vector_type(8)));

// ---------------- helpers ----------------
__device__ inline bf16_t tobf(float f) {
    union { float f; unsigned u; } in; in.f = f;
    unsigned u = in.u;
    unsigned r = u + 0x7FFFu + ((u >> 16) & 1u);   // round-to-nearest-even
    union { unsigned short s; bf16_t b; } out;
    out.s = (unsigned short)(r >> 16);
    return out.b;
}

__device__ inline float selu_f(float x) {
    const float a  = 1.6732632423543772f;
    const float sc = 1.0507009873554805f;
    return x > 0.f ? sc * x : sc * a * (__expf(x) - 1.f);
}

__device__ inline float wred_sum(float s) {
#pragma unroll
    for (int off = 16; off; off >>= 1) s += __shfl_xor(s, off, 32);
    return s;
}

__device__ inline void atomicMaxF(float* addr, float val) {
    int* ai = (int*)addr;
    int old = *ai;
    while (true) {
        float f = __int_as_float(old);
        if (f >= val) break;
        int assumed = old;
        old = atomicCAS(ai, assumed, __float_as_int(val));
        if (old == assumed) break;
    }
}

// ---- CDNA5 async global->LDS copy (ASYNCcnt-tracked; see CDNA5 ISA) ----
__device__ inline unsigned lds32(const void* p) {
    // generic -> LDS(as3) addrspacecast, then truncate to 32-bit LDS byte address
    return (unsigned)(uintptr_t)(const __attribute__((address_space(3))) char*)p;
}

__device__ inline void async_copy16(unsigned lds_addr, const void* gaddr) {
    asm volatile("global_load_async_to_lds_b128 %0, %1, off"
                 :: "v"(lds_addr), "v"((unsigned long long)(uintptr_t)gaddr)
                 : "memory");
}

__device__ inline void wait_async0() {
    asm volatile("s_wait_asynccnt 0x0" ::: "memory");
}

// ---------------- elementwise / prep kernels ----------------
__global__ void k_fill(float* p, float v, int n) {
    int i = blockIdx.x * blockDim.x + threadIdx.x;
    if (i < n) p[i] = v;
}

// h = mesh_feat @ w_in + b_in        (8192x4 @ 4x64)
__global__ void k_embed(const float* __restrict__ mf, const float* __restrict__ w,
                        const float* __restrict__ b, float* __restrict__ h) {
    int i = blockIdx.x * blockDim.x + threadIdx.x;
    if (i >= NN * EMB) return;
    int n = i >> 6, e = i & 63;
    float s = b[e];
#pragma unroll
    for (int d = 0; d < DIN; ++d) s += mf[n * DIN + d] * w[d * EMB + e];
    h[i] = s;
}

// fp32 (rows x K) -> bf16 (rows x Kp), zero padded
__global__ void k_tobf_pad(const float* __restrict__ src, bf16_t* __restrict__ dst,
                           int K, int Kp, int total) {
    int i = blockIdx.x * blockDim.x + threadIdx.x;
    if (i >= total) return;
    int r = i / Kp, j = i - r * Kp;
    dst[i] = (j < K) ? tobf(src[r * K + j]) : tobf(0.f);
}

// W (K x Nc) fp32 -> Wt (Nc x Kp) bf16 transposed, zero padded in K
__global__ void k_wt(const float* __restrict__ W, bf16_t* __restrict__ Wt,
                     int K, int Nc, int Kp) {
    int i = blockIdx.x * blockDim.x + threadIdx.x;
    if (i >= Nc * Kp) return;
    int n = i / Kp, kp = i - n * Kp;
    Wt[i] = (kp < K) ? tobf(W[kp * Nc + n]) : tobf(0.f);
}

// ---------------- WMMA GEMM v1 (direct global): wave tile 16 x (16*NT) ----------------
// ACT: 0=none 1=relu 2=selu
template <int NT, int ACT>
__global__ __launch_bounds__(32) void k_gemm(const bf16_t* __restrict__ A,
                                             const bf16_t* __restrict__ Wt,
                                             const float* __restrict__ bias,
                                             float* __restrict__ C,
                                             int Kp, int Nc) {
    const int lane = threadIdx.x & 31;
    const int m0 = blockIdx.x * 16;
    const int n0 = blockIdx.y * 16 * NT;
    const int row = m0 + (lane & 15);
    const int kloA = (lane >> 4) * 8;    // A: upper half-wave holds K+8..15 / K+24..31
    const int kbB  = (lane >> 4) * 8;    // B: upper half-wave holds K+8..15 / K+24..31

    union F { v16bf v; uint4 u[2]; };
    v8f acc[NT];
    const v8f zero = {0.f, 0.f, 0.f, 0.f, 0.f, 0.f, 0.f, 0.f};
#pragma unroll
    for (int t = 0; t < NT; ++t) acc[t] = zero;

    const bf16_t* arow = A + (size_t)row * Kp;
    for (int k = 0; k < Kp; k += 32) {
        F a;
        a.u[0] = *(const uint4*)(arow + k + kloA);
        a.u[1] = *(const uint4*)(arow + k + 16 + kloA);
#pragma unroll
        for (int t = 0; t < NT; ++t) {
            int col = n0 + t * 16 + (lane & 15);
            const bf16_t* bp = Wt + (size_t)col * Kp + k + kbB;
            F b;
            b.u[0] = *(const uint4*)(bp);
            b.u[1] = *(const uint4*)(bp + 16);
            acc[t] = __builtin_amdgcn_wmma_f32_16x16x32_bf16(
                false, a.v, false, b.v, (short)0, acc[t], false, false);
        }
    }

    const int half = (lane >> 4) * 8;   // C: lanes 16-31 hold M = v+8
#pragma unroll
    for (int t = 0; t < NT; ++t) {
        int col = n0 + t * 16 + (lane & 15);
        float bv = bias ? bias[col] : 0.f;
#pragma unroll
        for (int v = 0; v < 8; ++v) {
            float val = acc[t][v] + bv;
            if (ACT == 1) val = fmaxf(val, 0.f);
            else if (ACT == 2) val = selu_f(val);
            C[(size_t)(m0 + half + v) * Nc + col] = val;
        }
    }
}

// ---------------- WMMA GEMM v2: 4 waves / block, 64x64 block tile ----------------
// Double-buffered async global->LDS staging of 64x32 A and 64x32 B bf16 tiles.
// Per stage (8KB): A rows at smem[s][row*64 ..], B cols at smem[s][4096 + col*64 ..].
__global__ __launch_bounds__(128) void k_gemm_lds(const bf16_t* __restrict__ A,
                                                  const bf16_t* __restrict__ Wt,
                                                  float* __restrict__ C,
                                                  int Kp, int Nc) {
    __shared__ char smem[2][8192];
    const int tid  = threadIdx.x;
    const int lane = tid & 31;
    const int warp = tid >> 5;
    const int m0 = blockIdx.x * 64;
    const int n0 = blockIdx.y * 64;

    const size_t rowB = (size_t)Kp * 2;   // row stride in bytes (A and Wt share Kp)
    const char* Ab = (const char*)A  + (size_t)m0 * rowB;
    const char* Bb = (const char*)Wt + (size_t)n0 * rowB;

    // stage k-slice [k, k+32) into buffer s: 256 + 256 16B chunks, 4 async ops/thread
    auto stage = [&](int s, int k) {
        const size_t kb = (size_t)k * 2;
        for (int c = tid; c < 256; c += 128) {
            int row = c >> 2, o16 = (c & 3) * 16;
            async_copy16(lds32(&smem[s][row * 64 + o16]),
                         Ab + (size_t)row * rowB + kb + o16);
        }
        for (int c = tid; c < 256; c += 128) {
            int col = c >> 2, o16 = (c & 3) * 16;
            async_copy16(lds32(&smem[s][4096 + col * 64 + o16]),
                         Bb + (size_t)col * rowB + kb + o16);
        }
    };

    union F { v16bf v; uint4 u[2]; };
    v8f acc[4];
    const v8f zero = {0.f, 0.f, 0.f, 0.f, 0.f, 0.f, 0.f, 0.f};
#pragma unroll
    for (int t = 0; t < 4; ++t) acc[t] = zero;

    stage(0, 0);
    wait_async0();
    __syncthreads();

    const int nk = Kp / 32;
    const int arow_off = (16 * warp + (lane & 15)) * 64 + (lane >> 4) * 16;
    const int bsel = (lane >> 4) * 16;

    for (int i = 0; i < nk; ++i) {
        int s = i & 1;
        if (i + 1 < nk) stage(1 - s, (i + 1) * 32);   // overlap copy with compute

        F a;
        const char* ap = &smem[s][arow_off];
        a.u[0] = *(const uint4*)(ap);
        a.u[1] = *(const uint4*)(ap + 32);
#pragma unroll
        for (int t = 0; t < 4; ++t) {
            const char* bp = &smem[s][4096 + (t * 16 + (lane & 15)) * 64 + bsel];
            F b;
            b.u[0] = *(const uint4*)(bp);
            b.u[1] = *(const uint4*)(bp + 32);
            acc[t] = __builtin_amdgcn_wmma_f32_16x16x32_bf16(
                false, a.v, false, b.v, (short)0, acc[t], false, false);
        }
        wait_async0();        // own stage-(i+1) copies complete
        __syncthreads();      // everyone done reading buffer s before it is refilled
    }

    const int half = (lane >> 4) * 8;
#pragma unroll
    for (int t = 0; t < 4; ++t) {
        int col = n0 + t * 16 + (lane & 15);
#pragma unroll
        for (int v = 0; v < 8; ++v)
            C[(size_t)(m0 + 16 * warp + half + v) * Nc + col] = acc[t][v];
    }
}

// ---------------- attention: one wave per (b, h, query) ----------------
__global__ __launch_bounds__(32) void k_attn(const float* __restrict__ q,
                                             const float* __restrict__ k,
                                             const float* __restrict__ v,
                                             float* __restrict__ o) {
    const int idx = blockIdx.x;            // b*TH*S + h*S + s
    const int lane = threadIdx.x;
    const int s = idx & (SS - 1);
    const int h = (idx >> 10) & 3;
    const int b = idx >> 12;
    const int n = b * SS + s;

    float qd[HD];
    const float* qp = q + (size_t)n * EMB + h * HD;
#pragma unroll
    for (int d = 0; d < HD; ++d) qd[d] = qp[d];

    float mx = -1e30f, sum = 0.f;
    float oacc[HD];
#pragma unroll
    for (int d = 0; d < HD; ++d) oacc[d] = 0.f;

    for (int j = lane; j < SS; j += 32) {
        const float* kp = k + (size_t)(b * SS + j) * EMB + h * HD;
        float dot = 0.f;
#pragma unroll
        for (int d = 0; d < HD; ++d) dot += qd[d] * kp[d];
        dot *= 0.25f;                        // 1/sqrt(16)
        float nmx = fmaxf(mx, dot);
        float corr = __expf(mx - nmx);
        float p = __expf(dot - nmx);
        sum = sum * corr + p;
        const float* vp = v + (size_t)(b * SS + j) * EMB + h * HD;
#pragma unroll
        for (int d = 0; d < HD; ++d) oacc[d] = oacc[d] * corr + p * vp[d];
        mx = nmx;
    }
#pragma unroll
    for (int off = 16; off; off >>= 1) {
        float omx = __shfl_xor(mx, off, 32);
        float osm = __shfl_xor(sum, off, 32);
        float nmx = fmaxf(mx, omx);
        float c1 = __expf(mx - nmx), c2 = __expf(omx - nmx);
        sum = sum * c1 + osm * c2;
#pragma unroll
        for (int d = 0; d < HD; ++d) {
            float ov = __shfl_xor(oacc[d], off, 32);
            oacc[d] = oacc[d] * c1 + ov * c2;
        }
        mx = nmx;
    }
    float inv = 1.f / sum;
#pragma unroll
    for (int d = 0; d < HD; ++d)
        if (lane == d) o[(size_t)n * EMB + h * HD + d] = oacc[d] * inv;
}

// h = LayerNorm(h + add) * g + b    (row = 64, one wave, 2 elems/lane)
__global__ __launch_bounds__(32) void k_addln(float* __restrict__ h, const float* __restrict__ add,
                                              const float* __restrict__ g, const float* __restrict__ b) {
    int n = blockIdx.x, lane = threadIdx.x;
    float x0 = h[n * EMB + lane] + add[n * EMB + lane];
    float x1 = h[n * EMB + 32 + lane] + add[n * EMB + 32 + lane];
    float mu = wred_sum(x0 + x1) * (1.f / EMB);
    float d0 = x0 - mu, d1 = x1 - mu;
    float var = wred_sum(d0 * d0 + d1 * d1) * (1.f / EMB);
    float r = rsqrtf(var + 1e-5f);
    h[n * EMB + lane]      = d0 * r * g[lane] + b[lane];
    h[n * EMB + 32 + lane] = d1 * r * g[32 + lane] + b[32 + lane];
}

// feats_bf[n, 0..31] = bf16(concat(x[n,2:7], tfeat[n,0:16]), pad)
__global__ void k_feats(const float* __restrict__ x, const float* __restrict__ tf,
                        bf16_t* __restrict__ out) {
    int i = blockIdx.x * blockDim.x + threadIdx.x;
    if (i >= NN * FKP) return;
    int n = i / FKP, j = i - n * FKP;
    float v = 0.f;
    if (j < 5) v = x[n * 7 + 2 + j];
    else if (j < FEATC) v = tf[n * TOUT + (j - 5)];
    out[i] = tobf(v);
}

__global__ void k_phi0(const float* __restrict__ mf, float* __restrict__ phi) {
    int n = blockIdx.x * blockDim.x + threadIdx.x;
    if (n < NN) phi[n] = mf[n * DIN + 3];
}

// infeat_bf[n, 0..543] = bf16(concat(phi[n], hidden[n,:]), pad)
__global__ void k_infeat(const float* __restrict__ phi, const float* __restrict__ hid,
                         bf16_t* __restrict__ out) {
    int i = blockIdx.x * blockDim.x + threadIdx.x;
    if (i >= NN * INFKP) return;
    int n = i / INFKP, j = i - n * INFKP;
    float v = 0.f;
    if (j == 0) v = phi[n];
    else if (j <= HID) v = hid[n * HID + (j - 1)];
    out[i] = tobf(v);
}

// score[e,h] = sum_c leakyrelu(xl[src,h,c]+xr[dst,h,c]) * att[h,c]; atomic seg-max into m
__global__ void k_edge_score(const int* __restrict__ ei, const float* __restrict__ xl,
                             const float* __restrict__ xr, const float* __restrict__ att,
                             float* __restrict__ score, float* __restrict__ m) {
    int w = blockIdx.x * (blockDim.x >> 5) + (threadIdx.x >> 5);
    int lane = threadIdx.x & 31;
    if (w >= E2 * GH) return;
    int e = w / GH, h = w - e * GH;
    int src, dst;
    if (e < EE) { src = ei[e]; dst = ei[EE + e]; } else { src = dst = e - EE; }
    const float* xlp = xl + (size_t)src * GCOL + h * HID;
    const float* xrp = xr + (size_t)dst * GCOL + h * HID;
    const float* ap = att + h * HID;
    float s = 0.f;
    for (int c = lane; c < HID; c += 32) {
        float t = xlp[c] + xrp[c];
        t = t > 0.f ? t : 0.2f * t;
        s += t * ap[c];
    }
    s = wred_sum(s);
    if (lane == 0) {
        score[(size_t)e * GH + h] = s;
        atomicMaxF(&m[dst * GH + h], s);
    }
}

// score <- exp(score - m[dst]); denom[dst] += score
__global__ void k_edge_exp(const int* __restrict__ ei, float* __restrict__ score,
                           const float* __restrict__ m, float* __restrict__ denom) {
    int i = blockIdx.x * blockDim.x + threadIdx.x;
    if (i >= E2 * GH) return;
    int e = i / GH, h = i - e * GH;
    int dst = (e < EE) ? ei[EE + e] : (e - EE);
    float ex = __expf(score[i] - m[dst * GH + h]);
    score[i] = ex;
    atomicAdd(&denom[dst * GH + h], ex);
}

// g[dst, c] += (alpha/GH) * xl[src, h, c]; 4 channels per thread
__global__ void k_edge_agg(const int* __restrict__ ei, const float* __restrict__ ex,
                           const float* __restrict__ denom, const float* __restrict__ xl,
                           float* __restrict__ g) {
    int i = blockIdx.x * blockDim.x + threadIdx.x;
    if (i >= E2 * GH * (HID / 4)) return;
    int c4 = i & (HID / 4 - 1);
    int r = i >> 7;               // HID/4 == 128
    int h = r % GH, e = r / GH;
    int src, dst;
    if (e < EE) { src = ei[e]; dst = ei[EE + e]; } else { src = dst = e - EE; }
    float alpha = ex[(size_t)e * GH + h] / (denom[dst * GH + h] + 1e-16f);
    float w = alpha * (1.f / GH);
    const float4 xv = *(const float4*)(xl + (size_t)src * GCOL + h * HID + c4 * 4);
    float* gp = g + (size_t)dst * HID + c4 * 4;
    atomicAdd(gp + 0, w * xv.x);
    atomicAdd(gp + 1, w * xv.y);
    atomicAdd(gp + 2, w * xv.z);
    atomicAdd(gp + 3, w * xv.w);
}

// hidden = selu(g + g_bias)
__global__ void k_selu_bias(const float* __restrict__ g, const float* __restrict__ gb,
                            float* __restrict__ hid) {
    int i = blockIdx.x * blockDim.x + threadIdx.x;
    if (i >= NN * HID) return;
    hid[i] = selu_f(g[i] + gb[i & (HID - 1)]);
}

// phi[n] = hidden[n,:] . w_out + b_out   (wave per row)
__global__ __launch_bounds__(32) void k_phi_out(const float* __restrict__ hid,
                                                const float* __restrict__ wo,
                                                const float* __restrict__ bo,
                                                float* __restrict__ phi) {
    int n = blockIdx.x, lane = threadIdx.x;
    float s = 0.f;
    for (int c = lane; c < HID; c += 32) s += hid[(size_t)n * HID + c] * wo[c];
    s = wred_sum(s);
    if (lane == 0) phi[n] = s + bo[0];
}

// ---------------- host launch ----------------
extern "C" void kernel_launch(void* const* d_in, const int* in_sizes, int n_in,
                              void* d_out, int out_size, void* d_ws, size_t ws_size,
                              hipStream_t stream) {
    const float* x      = (const float*)d_in[0];
    const float* mesh   = (const float*)d_in[1];
    const int*   ei     = (const int*)d_in[3];
    const float* w_in   = (const float*)d_in[4];
    const float* b_in   = (const float*)d_in[5];
    const float* wq     = (const float*)d_in[6];
    const float* bq     = (const float*)d_in[7];
    const float* wk     = (const float*)d_in[8];
    const float* bk     = (const float*)d_in[9];
    const float* wv     = (const float*)d_in[10];
    const float* bv     = (const float*)d_in[11];
    const float* wo     = (const float*)d_in[12];
    const float* bo     = (const float*)d_in[13];
    const float* ln1g   = (const float*)d_in[14];
    const float* ln1b   = (const float*)d_in[15];
    const float* wff1   = (const float*)d_in[16];
    const float* bff1   = (const float*)d_in[17];
    const float* wff2   = (const float*)d_in[18];
    const float* bff2   = (const float*)d_in[19];
    const float* ln2g   = (const float*)d_in[20];
    const float* ln2b   = (const float*)d_in[21];
    const float* wtout  = (const float*)d_in[22];
    const float* btout  = (const float*)d_in[23];
    const float* wlin   = (const float*)d_in[24];
    const float* blin   = (const float*)d_in[25];
    const float* wl     = (const float*)d_in[26];
    const float* wr     = (const float*)d_in[27];
    const float* att    = (const float*)d_in[28];
    const float* gbias  = (const float*)d_in[29];
    const float* w_out  = (const float*)d_in[30];
    const float* b_out  = (const float*)d_in[31];

    char* base = (char*)d_ws;
    size_t off = 0;
    auto alloc = [&](size_t bytes) -> void* {
        void* p = base + off;
        off = (off + bytes + 255) & ~(size_t)255;
        return p;
    };
    float*  h        = (float*)alloc((size_t)NN * EMB * 4);
    float*  qb       = (float*)alloc((size_t)NN * EMB * 4);
    float*  kb       = (float*)alloc((size_t)NN * EMB * 4);
    float*  vb       = (float*)alloc((size_t)NN * EMB * 4);
    float*  ob       = (float*)alloc((size_t)NN * EMB * 4);
    float*  tmp64    = (float*)alloc((size_t)NN * EMB * 4);
    float*  ff1      = (float*)alloc((size_t)NN * DFF * 4);
    float*  tfeat    = (float*)alloc((size_t)NN * TOUT * 4);
    bf16_t* h_bf     = (bf16_t*)alloc((size_t)NN * EMB * 2);
    bf16_t* o_bf     = (bf16_t*)alloc((size_t)NN * EMB * 2);
    bf16_t* ff1_bf   = (bf16_t*)alloc((size_t)NN * DFF * 2);
    bf16_t* feats_bf = (bf16_t*)alloc((size_t)NN * FKP * 2);
    bf16_t* inf_bf   = (bf16_t*)alloc((size_t)NN * INFKP * 2);
    float*  hidden   = (float*)alloc((size_t)NN * HID * 4);
    float*  gacc     = (float*)alloc((size_t)NN * HID * 4);
    float*  xl       = (float*)alloc((size_t)NN * GCOL * 4);
    float*  xr       = (float*)alloc((size_t)NN * GCOL * 4);
    float*  score    = (float*)alloc((size_t)E2 * GH * 4);
    float*  mseg     = (float*)alloc((size_t)NN * GH * 4);
    float*  dseg     = (float*)alloc((size_t)NN * GH * 4);
    float*  phi      = (float*)alloc((size_t)NN * 4);
    bf16_t* wq_t     = (bf16_t*)alloc((size_t)EMB * EMB * 2);
    bf16_t* wk_t     = (bf16_t*)alloc((size_t)EMB * EMB * 2);
    bf16_t* wv_t     = (bf16_t*)alloc((size_t)EMB * EMB * 2);
    bf16_t* wo_t     = (bf16_t*)alloc((size_t)EMB * EMB * 2);
    bf16_t* wff1_t   = (bf16_t*)alloc((size_t)DFF * EMB * 2);
    bf16_t* wff2_t   = (bf16_t*)alloc((size_t)EMB * DFF * 2);
    bf16_t* wtout_t  = (bf16_t*)alloc((size_t)TOUT * EMB * 2);
    bf16_t* wlin_t   = (bf16_t*)alloc((size_t)HID * FKP * 2);
    bf16_t* wl_t     = (bf16_t*)alloc((size_t)GCOL * INFKP * 2);
    bf16_t* wr_t     = (bf16_t*)alloc((size_t)GCOL * INFKP * 2);

    const int TPB = 256;
    auto blk = [](long n, int t) { return dim3((unsigned)((n + t - 1) / t)); };

    // ---- weight prep (fp32 -> transposed, K-padded bf16) ----
    k_wt<<<blk((long)EMB * EMB, TPB), TPB, 0, stream>>>(wq, wq_t, EMB, EMB, EMB);
    k_wt<<<blk((long)EMB * EMB, TPB), TPB, 0, stream>>>(wk, wk_t, EMB, EMB, EMB);
    k_wt<<<blk((long)EMB * EMB, TPB), TPB, 0, stream>>>(wv, wv_t, EMB, EMB, EMB);
    k_wt<<<blk((long)EMB * EMB, TPB), TPB, 0, stream>>>(wo, wo_t, EMB, EMB, EMB);
    k_wt<<<blk((long)DFF * EMB, TPB), TPB, 0, stream>>>(wff1, wff1_t, EMB, DFF, EMB);
    k_wt<<<blk((long)EMB * DFF, TPB), TPB, 0, stream>>>(wff2, wff2_t, DFF, EMB, DFF);
    k_wt<<<blk((long)TOUT * EMB, TPB), TPB, 0, stream>>>(wtout, wtout_t, EMB, TOUT, EMB);
    k_wt<<<blk((long)HID * FKP, TPB), TPB, 0, stream>>>(wlin, wlin_t, FEATC, HID, FKP);
    k_wt<<<blk((long)GCOL * INFKP, TPB), TPB, 0, stream>>>(wl, wl_t, INFK, GCOL, INFKP);
    k_wt<<<blk((long)GCOL * INFKP, TPB), TPB, 0, stream>>>(wr, wr_t, INFK, GCOL, INFKP);

    // ---- transformer ----
    k_embed<<<blk((long)NN * EMB, TPB), TPB, 0, stream>>>(mesh, w_in, b_in, h);
    k_tobf_pad<<<blk((long)NN * EMB, TPB), TPB, 0, stream>>>(h, h_bf, EMB, EMB, NN * EMB);

    dim3 g64(NN / 16, EMB / 64);
    k_gemm<4, 0><<<g64, 32, 0, stream>>>(h_bf, wq_t, bq, qb, EMB, EMB);
    k_gemm<4, 0><<<g64, 32, 0, stream>>>(h_bf, wk_t, bk, kb, EMB, EMB);
    k_gemm<4, 0><<<g64, 32, 0, stream>>>(h_bf, wv_t, bv, vb, EMB, EMB);

    k_attn<<<BB * THD * SS, 32, 0, stream>>>(qb, kb, vb, ob);

    k_tobf_pad<<<blk((long)NN * EMB, TPB), TPB, 0, stream>>>(ob, o_bf, EMB, EMB, NN * EMB);
    k_gemm<4, 0><<<g64, 32, 0, stream>>>(o_bf, wo_t, bo, tmp64, EMB, EMB);
    k_addln<<<NN, 32, 0, stream>>>(h, tmp64, ln1g, ln1b);

    k_tobf_pad<<<blk((long)NN * EMB, TPB), TPB, 0, stream>>>(h, h_bf, EMB, EMB, NN * EMB);
    dim3 g256(NN / 16, DFF / 64);
    k_gemm<4, 1><<<g256, 32, 0, stream>>>(h_bf, wff1_t, bff1, ff1, EMB, DFF);   // relu
    k_tobf_pad<<<blk((long)NN * DFF, TPB), TPB, 0, stream>>>(ff1, ff1_bf, DFF, DFF, NN * DFF);
    k_gemm<4, 0><<<g64, 32, 0, stream>>>(ff1_bf, wff2_t, bff2, tmp64, DFF, EMB);
    k_addln<<<NN, 32, 0, stream>>>(h, tmp64, ln2g, ln2b);

    k_tobf_pad<<<blk((long)NN * EMB, TPB), TPB, 0, stream>>>(h, h_bf, EMB, EMB, NN * EMB);
    dim3 g16(NN / 16, TOUT / 16);
    k_gemm<1, 0><<<g16, 32, 0, stream>>>(h_bf, wtout_t, btout, tfeat, EMB, TOUT);

    // ---- features + first hidden ----
    k_feats<<<blk((long)NN * FKP, TPB), TPB, 0, stream>>>(x, tfeat, feats_bf);
    dim3 g512(NN / 16, HID / 64);
    k_gemm<4, 2><<<g512, 32, 0, stream>>>(feats_bf, wlin_t, blin, hidden, FKP, HID);  // selu
    k_phi0<<<blk(NN, TPB), TPB, 0, stream>>>(mesh, phi);

    // ---- GATv2 loops ----
    dim3 gbig(NN / 64, GCOL / 64);       // async-LDS WMMA GEMM, 64x64 tiles
    const float* phi_cur = phi;
    for (int it = 0; it < NUM_LOOP; ++it) {
        k_infeat<<<blk((long)NN * INFKP, TPB), TPB, 0, stream>>>(phi_cur, hidden, inf_bf);
        k_gemm_lds<<<gbig, 128, 0, stream>>>(inf_bf, wl_t, xl, INFKP, GCOL);
        k_gemm_lds<<<gbig, 128, 0, stream>>>(inf_bf, wr_t, xr, INFKP, GCOL);

        k_fill<<<blk((long)NN * GH, TPB), TPB, 0, stream>>>(mseg, -1e30f, NN * GH);
        k_fill<<<blk((long)NN * GH, TPB), TPB, 0, stream>>>(dseg, 0.f, NN * GH);
        k_fill<<<blk((long)NN * HID, TPB), TPB, 0, stream>>>(gacc, 0.f, NN * HID);

        long nwaves = (long)E2 * GH;
        k_edge_score<<<blk(nwaves * 32, TPB), TPB, 0, stream>>>(ei, xl, xr, att, score, mseg);
        k_edge_exp<<<blk(nwaves, TPB), TPB, 0, stream>>>(ei, score, mseg, dseg);
        k_edge_agg<<<blk(nwaves * (HID / 4), TPB), TPB, 0, stream>>>(ei, score, dseg, xl, gacc);

        k_selu_bias<<<blk((long)NN * HID, TPB), TPB, 0, stream>>>(gacc, gbias, hidden);

        float* phi_dst = (it == NUM_LOOP - 1) ? (float*)d_out : phi;
        k_phi_out<<<NN, 32, 0, stream>>>(hidden, w_out, b_out, phi_dst);
        phi_cur = phi_dst;
    }
}
